// EdgeAwareMultiHeadAttention_249108103244
// MI455X (gfx1250) — matrix operand order repaired and checked by reading, not compiled, not run on hardware
//
#include <hip/hip_runtime.h>
#include <hip/hip_bf16.h>

// ---------------------------------------------------------------------------
// EdgeAwareMultiHeadAttention for MI455X (gfx1250, wave32, WMMA bf16)
// B=2, N=2048, E=512, H=8, D=64, EDGE_DIM=16
// ---------------------------------------------------------------------------

#define BB 2
#define NN 2048
#define EE 512
#define HH 8
#define DD 64
#define ED 16

typedef __bf16 bf16x2  __attribute__((ext_vector_type(2)));
typedef __bf16 bf16x8  __attribute__((ext_vector_type(8)));
typedef __bf16 bf16x16 __attribute__((ext_vector_type(16)));
typedef float  f32x8   __attribute__((ext_vector_type(8)));

// ---- f32 -> bf16 conversion: prefer HW packed convert, fall back to RNE bit math
#if __has_builtin(__builtin_amdgcn_cvt_pk_bf16_f32)
__device__ __forceinline__ bf16x2 f2bf2(float a, float b) {
    return __builtin_bit_cast(bf16x2, __builtin_amdgcn_cvt_pk_bf16_f32(a, b));
}
__device__ __forceinline__ __bf16 f2bf(float f) { return f2bf2(f, f)[0]; }
#else
__device__ __forceinline__ __bf16 f2bf(float f) {
    unsigned u = __builtin_bit_cast(unsigned, f);
    unsigned r = u + 0x7FFFu + ((u >> 16) & 1u);
    unsigned short h = (unsigned short)(r >> 16);
    return __builtin_bit_cast(__bf16, h);
}
__device__ __forceinline__ bf16x2 f2bf2(float a, float b) {
    bf16x2 t; t[0] = f2bf(a); t[1] = f2bf(b); return t;
}
#endif

__device__ __forceinline__ bf16x16 bfcat(bf16x8 lo, bf16x8 hi) {
    return __builtin_shufflevector(lo, hi, 0,1,2,3,4,5,6,7,8,9,10,11,12,13,14,15);
}

// 16 contiguous bf16 (32B, 16B-aligned at all call sites)
__device__ __forceinline__ bf16x16 frag16(const __bf16* p) {
    const bf16x8* q = (const bf16x8*)p;
    return bfcat(q[0], q[1]);
}
// two 8-element contiguous chunks (A-fragment pattern)
__device__ __forceinline__ bf16x16 frag2(const __bf16* lo, const __bf16* hi) {
    return bfcat(*(const bf16x8*)lo, *(const bf16x8*)hi);
}

// A-fragment (16x32 bf16, M=lane%16, K = kbase + 16g contiguous) from f32 row
__device__ __forceinline__ bf16x16 a_frag_f32(const float* row, int kbase, int g) {
    bf16x16 a;
    const float* p0 = row + kbase + 16 * g;
    const float* p1 = row + kbase + 16 * g + 8;
#pragma unroll
    for (int i = 0; i < 4; ++i) {
        bf16x2 t = f2bf2(p0[2 * i], p0[2 * i + 1]);
        a[2 * i] = t[0]; a[2 * i + 1] = t[1];
    }
#pragma unroll
    for (int i = 0; i < 4; ++i) {
        bf16x2 t = f2bf2(p1[2 * i], p1[2 * i + 1]);
        a[8 + 2 * i] = t[0]; a[9 + 2 * i] = t[1];
    }
    return a;
}
// B-fragment (32x16 bf16, col=lane%16, K = e + 16g contiguous) from f32 row
__device__ __forceinline__ bf16x16 b_frag_f32(const float* p) {
    bf16x16 v;
#pragma unroll
    for (int i = 0; i < 8; ++i) {
        bf16x2 t = f2bf2(p[2 * i], p[2 * i + 1]);
        v[2 * i] = t[0]; v[2 * i + 1] = t[1];
    }
    return v;
}

#define WMMA_BF16(A, Bf, C) \
    __builtin_amdgcn_wmma_f32_16x16x32_bf16(false, (A), false, (Bf), (short)0, (C), false, false)

// ---------------------------------------------------------------------------
// Kernel 1: Y = X @ W^T + b   (4096 x 512 x 512), bf16 WMMA, f32 accum.
// VMODE=0: write (b,h,n,d) row-major bf16   (Q, K)
// VMODE=1: write (b,h,d,n) transposed bf16  (V, so PV B-fragments are contiguous)
// ---------------------------------------------------------------------------
template <int VMODE>
__global__ __launch_bounds__(32) void proj_kernel(
    const float* __restrict__ x, const float* __restrict__ W,
    const float* __restrict__ bias, unsigned short* __restrict__ outp)
{
    const int lane = threadIdx.x, g = lane >> 4, ln = lane & 15;
    const int col0 = blockIdx.x * 16, row0 = blockIdx.y * 16;
    const float* xrow = x + (size_t)(row0 + ln) * EE;
    const float* wrow = W + (size_t)(col0 + ln) * EE;

    f32x8 acc = {0.f, 0.f, 0.f, 0.f, 0.f, 0.f, 0.f, 0.f};
    for (int k0 = 0; k0 < EE; k0 += 32) {
        bf16x16 a  = a_frag_f32(xrow, k0, g);
        bf16x16 bf = b_frag_f32(wrow + k0 + 16 * g);
        acc = WMMA_BF16(a, bf, acc);
    }

    const int col = col0 + ln;
    const float bv = bias[col];
    const int hh = col >> 6, dd = col & 63;
    __bf16* out = reinterpret_cast<__bf16*>(outp);
#pragma unroll
    for (int r = 0; r < 8; ++r) {
        const int i  = row0 + r + 8 * g;       // global row in [0, 4096)
        const int bi = i >> 11, ni = i & 2047;
        const float v = acc[r] + bv;
        size_t idx;
        if (VMODE) idx = (size_t)((bi * HH + hh) * DD + dd) * NN + ni;
        else       idx = (size_t)((bi * HH + hh) * NN + ni) * DD + dd;
        out[idx] = f2bf(v);
    }
}

// ---------------------------------------------------------------------------
// Kernel 2: fused edge-MLP + streaming-softmax attention.
// One wave per (b, h, 16-row query tile); streams key/value in 32-col tiles.
// ---------------------------------------------------------------------------
__global__ __launch_bounds__(32) void attn_kernel(
    const unsigned short* __restrict__ q_u, const unsigned short* __restrict__ k_u,
    const unsigned short* __restrict__ vt_u, const float* __restrict__ adj,
    const float* __restrict__ We1, const float* __restrict__ be1,
    const float* __restrict__ We2, const float* __restrict__ be2,
    const float* __restrict__ ebias, unsigned short* __restrict__ ao_u)
{
    const int lane = threadIdx.x, g = lane >> 4, ln = lane & 15;
    const int row0 = blockIdx.x * 16, h = blockIdx.y, b = blockIdx.z;

    const __bf16* Q = ((const __bf16*)q_u)  + (size_t)(b * HH + h) * NN * DD;
    const __bf16* K = ((const __bf16*)k_u)  + (size_t)(b * HH + h) * NN * DD;
    const __bf16* V = ((const __bf16*)vt_u) + (size_t)(b * HH + h) * DD * NN;
    const float*  A = adj + (size_t)b * NN * NN;

    // Q A-fragments for d-chunks [0,32) and [32,64)
    const __bf16* qrow = Q + (size_t)(row0 + ln) * DD;
    const bf16x16 qa0 = frag2(qrow +      16 * g, qrow +  8 + 16 * g);
    const bf16x16 qa1 = frag2(qrow + 32 + 16 * g, qrow + 40 + 16 * g);

    // edge-MLP constants (uniform)
    float we1[ED], be1r[ED], we2[ED];
#pragma unroll
    for (int e = 0; e < ED; ++e) {
        we1[e]  = We1[e];
        be1r[e] = be1[e];
        we2[e]  = We2[h * ED + e];
    }
    const float be2r = be2[h];
    const float eb   = ebias[h];

    float mrun[8], lrun[8];
    f32x8 o0 = {0.f,0.f,0.f,0.f,0.f,0.f,0.f,0.f};
    f32x8 o1 = o0, o2 = o0, o3 = o0;
#pragma unroll
    for (int r = 0; r < 8; ++r) { mrun[r] = -3.0e38f; lrun[r] = 0.f; }

    __shared__ __align__(16) __bf16 plds[16 * 32];

#pragma unroll 1
    for (int j0 = 0; j0 < NN; j0 += 32) {
        // S tiles: cols [j0, j0+16) and [j0+16, j0+32), K-dim = d (64 = 2 chunks)
        f32x8 s0 = {0.f,0.f,0.f,0.f,0.f,0.f,0.f,0.f};
        f32x8 s1 = s0;
        {
            const __bf16* kr0 = K + (size_t)(j0 + ln) * DD;
            const __bf16* kr1 = K + (size_t)(j0 + 16 + ln) * DD;
            s0 = WMMA_BF16(qa0, frag16(kr0 + 16 * g),      s0);
            s0 = WMMA_BF16(qa1, frag16(kr0 + 32 + 16 * g), s0);
            s1 = WMMA_BF16(qa0, frag16(kr1 + 16 * g),      s1);
            s1 = WMMA_BF16(qa1, frag16(kr1 + 32 + 16 * g), s1);
        }

        // fuse edge MLP + online softmax (C layout: row = r + 8g, col = ln)
#pragma unroll
        for (int r = 0; r < 8; ++r) {
            const float* arow = A + (size_t)(row0 + r + 8 * g) * NN + j0;
            // prefetch next j-tile of this adjacency row (global_prefetch_b8)
            if (j0 + 32 < NN) __builtin_prefetch(arow + 32 + ln, 0, 1);
            const float a0 = arow[ln], a1 = arow[16 + ln];
            float e0 = be2r, e1 = be2r;
#pragma unroll
            for (int e = 0; e < ED; ++e) {
                const float h0 = fmaxf(fmaf(a0, we1[e], be1r[e]), 0.f);
                const float h1 = fmaxf(fmaf(a1, we1[e], be1r[e]), 0.f);
                e0 = fmaf(h0, we2[e], e0);
                e1 = fmaf(h1, we2[e], e1);
            }
            const float sv0 = s0[r] * 0.125f + e0 + a0 * eb;   // scale = 64^-0.5
            const float sv1 = s1[r] * 0.125f + e1 + a1 * eb;

            float mx = fmaxf(sv0, sv1);
#pragma unroll
            for (int off = 1; off < 16; off <<= 1)
                mx = fmaxf(mx, __shfl_xor(mx, off, 32));
            const float mn   = fmaxf(mrun[r], mx);
            const float corr = __expf(mrun[r] - mn);
            mrun[r] = mn;
            const float p0 = __expf(sv0 - mn);
            const float p1 = __expf(sv1 - mn);
            float rs = p0 + p1;
#pragma unroll
            for (int off = 1; off < 16; off <<= 1)
                rs += __shfl_xor(rs, off, 32);
            lrun[r] = lrun[r] * corr + rs;
            o0[r] *= corr; o1[r] *= corr; o2[r] *= corr; o3[r] *= corr;

            plds[(r + 8 * g) * 32 + ln]      = f2bf(p0);
            plds[(r + 8 * g) * 32 + 16 + ln] = f2bf(p1);
        }

        // in-wave LDS store->load hazard: drain DScnt before A-fragment reload
        asm volatile("s_wait_dscnt 0" ::: "memory");

        // P as A-fragment (16x32 bf16 row-major in LDS)
        const bf16x16 pa = frag2(&plds[ln * 32 + 16 * g], &plds[ln * 32 + 8 + 16 * g]);

        // V B-fragments: V stored (d, n) transposed -> contiguous per lane
        o0 = WMMA_BF16(pa, frag16(V + (size_t)(ln)      * NN + j0 + 16 * g), o0);
        o1 = WMMA_BF16(pa, frag16(V + (size_t)(16 + ln) * NN + j0 + 16 * g), o1);
        o2 = WMMA_BF16(pa, frag16(V + (size_t)(32 + ln) * NN + j0 + 16 * g), o2);
        o3 = WMMA_BF16(pa, frag16(V + (size_t)(48 + ln) * NN + j0 + 16 * g), o3);
    }

    // epilogue: normalize and write AO in (b, n, h*64+d) row-major bf16
    __bf16* AO = (__bf16*)ao_u;
#pragma unroll
    for (int r = 0; r < 8; ++r) {
        const float inv = 1.0f / lrun[r];
        const int i = row0 + r + 8 * g;
        __bf16* orow = AO + (size_t)(b * NN + i) * EE + h * DD;
        orow[ln]      = f2bf(o0[r] * inv);
        orow[16 + ln] = f2bf(o1[r] * inv);
        orow[32 + ln] = f2bf(o2[r] * inv);
        orow[48 + ln] = f2bf(o3[r] * inv);
    }
}

// ---------------------------------------------------------------------------
// Kernel 3: out = AO @ Wo^T + bo  (bf16 A from workspace, f32 output)
// ---------------------------------------------------------------------------
__global__ __launch_bounds__(32) void oproj_kernel(
    const unsigned short* __restrict__ ao_u, const float* __restrict__ Wo,
    const float* __restrict__ bo, float* __restrict__ out)
{
    const int lane = threadIdx.x, g = lane >> 4, ln = lane & 15;
    const int col0 = blockIdx.x * 16, row0 = blockIdx.y * 16;
    const __bf16* arow = ((const __bf16*)ao_u) + (size_t)(row0 + ln) * EE;
    const float*  wrow = Wo + (size_t)(col0 + ln) * EE;

    f32x8 acc = {0.f,0.f,0.f,0.f,0.f,0.f,0.f,0.f};
    for (int k0 = 0; k0 < EE; k0 += 32) {
        bf16x16 a  = frag2(arow + k0 + 16 * g, arow + k0 + 8 + 16 * g);
        bf16x16 bf = b_frag_f32(wrow + k0 + 16 * g);
        acc = WMMA_BF16(a, bf, acc);
    }

    const int col = col0 + ln;
    const float bv = bo[col];
#pragma unroll
    for (int r = 0; r < 8; ++r) {
        const int i = row0 + r + 8 * g;
        out[(size_t)i * EE + col] = acc[r] + bv;
    }
}

// ---------------------------------------------------------------------------
extern "C" void kernel_launch(void* const* d_in, const int* in_sizes, int n_in,
                              void* d_out, int out_size, void* d_ws, size_t ws_size,
                              hipStream_t stream) {
    const float* x     = (const float*)d_in[0];
    const float* adj   = (const float*)d_in[1];
    const float* Wq    = (const float*)d_in[2];
    const float* bq    = (const float*)d_in[3];
    const float* Wk    = (const float*)d_in[4];
    const float* bk    = (const float*)d_in[5];
    const float* Wv    = (const float*)d_in[6];
    const float* bv    = (const float*)d_in[7];
    const float* Wo    = (const float*)d_in[8];
    const float* bo    = (const float*)d_in[9];
    const float* We1   = (const float*)d_in[10];
    const float* be1   = (const float*)d_in[11];
    const float* We2   = (const float*)d_in[12];
    const float* be2   = (const float*)d_in[13];
    const float* ebias = (const float*)d_in[14];

    const size_t SEG = (size_t)BB * HH * NN * DD;   // 2,097,152 bf16 elems = 4 MB
    unsigned short* qb = (unsigned short*)d_ws;
    unsigned short* kb = qb + SEG;
    unsigned short* vt = kb + SEG;
    unsigned short* ao = vt + SEG;

    dim3 gp(EE / 16, (BB * NN) / 16);   // 32 x 256 tiles
    dim3 bp(32);
    hipLaunchKernelGGL((proj_kernel<0>), gp, bp, 0, stream, x, Wq, bq, qb);
    hipLaunchKernelGGL((proj_kernel<0>), gp, bp, 0, stream, x, Wk, bk, kb);
    hipLaunchKernelGGL((proj_kernel<1>), gp, bp, 0, stream, x, Wv, bv, vt);

    dim3 ga(NN / 16, HH, BB);           // 128 x 8 x 2
    hipLaunchKernelGGL(attn_kernel, ga, bp, 0, stream,
                       qb, kb, vt, adj, We1, be1, We2, be2, ebias, ao);

    hipLaunchKernelGGL(oproj_kernel, gp, bp, 0, stream, ao, Wo, bo, (float*)d_out);
}
